// LongcatFlashExperts_43954695308102
// MI455X (gfx1250) — matrix operand order repaired and checked by reading, not compiled, not run on hardware
//
#include <hip/hip_runtime.h>
#include <hip/hip_bf16.h>
#include <stdint.h>

// ---- problem constants (match reference) ----
#define T_TOK 4096
#define H_DIM 1024
#define I_DIM 512
#define R_EXP 32
#define K_TOP 4
#define CAPS  768

typedef __bf16 bf16;
typedef __attribute__((ext_vector_type(16))) __bf16 v16bf;
typedef __attribute__((ext_vector_type(8)))  float  v8f;

// gfx1250 async global->LDS copy (ASYNCcnt path); fall back if unavailable.
// Param type per hipcc diagnostic: (int __vector_size__(16) AS1*, ... AS3*, imm, imm)
#if __has_builtin(__builtin_amdgcn_global_load_async_to_lds_b128) && \
    __has_builtin(__builtin_amdgcn_s_wait_asynccnt)
#define USE_ASYNC_LDS 1
typedef int v4i __attribute__((vector_size(4 * sizeof(int))));
typedef __attribute__((address_space(1))) v4i gv4i;
typedef __attribute__((address_space(3))) v4i lv4i;
#else
#define USE_ASYNC_LDS 0
#endif

// ---- workspace layout (bytes) ----
#define OFF_CNT 0                       // R_EXP ints
#define OFF_TOK 512                     // R_EXP*CAPS ints   (98304 B)
#define OFF_SW  (512 + 98304)           // R_EXP*CAPS floats (98304 B)
#define OFF_ZW  (512 + 2*98304)         // T_TOK floats      (16384 B)
#define OFF_MID (512 + 2*98304 + 16384) // R_EXP*CAPS*I_DIM bf16 (25165824 B)

// ------------------------------------------------------------------
// Kernel 1: routing — claim capacity slots, compute zero-expert weight
// ------------------------------------------------------------------
__global__ __launch_bounds__(256) void route_kernel(
    const long long* __restrict__ topk_idx, const float* __restrict__ topk_w,
    int* __restrict__ cnt, int* __restrict__ tok, float* __restrict__ sw,
    float* __restrict__ zw) {
  int t = blockIdx.x * 256 + threadIdx.x;
  if (t >= T_TOK) return;
  float z = 0.f;
  #pragma unroll
  for (int k = 0; k < K_TOP; ++k) {
    int e = (int)topk_idx[t * K_TOP + k];
    float w = topk_w[t * K_TOP + k];
    if (e >= R_EXP) {
      z += w;
    } else {
      int p = atomicAdd(&cnt[e], 1);
      if (p < CAPS) {
        tok[e * CAPS + p] = t;
        sw[e * CAPS + p] = w;
      }
    }
  }
  zw[t] = z;
}

// ------------------------------------------------------------------
// Kernel 2: weighted-identity init for zero experts (inits d_out)
// ------------------------------------------------------------------
__global__ __launch_bounds__(256) void zinit_kernel(
    const float* __restrict__ hs, const float* __restrict__ zw,
    float* __restrict__ out) {
  int i = blockIdx.x * 256 + threadIdx.x;
  out[i] = hs[i] * zw[i >> 10];   // H_DIM == 1024
}

// ------------------------------------------------------------------
// Kernel 3: grouped GEMM1 (x @ gate_up^T) + SiLU*up fused, bf16 WMMA
// grid: (I_DIM/128, CAPS/64, R_EXP), block 256 (8 waves).
// wave w: gate cols [bx*128+w*16, +16) and up cols (+I_DIM),
//         4 m-subtiles of 16 rows -> 8 accumulators, B reused 4x.
// ------------------------------------------------------------------
__global__ __launch_bounds__(256) void gemm1_kernel(
    const float* __restrict__ hs, const float* __restrict__ gup,
    const int* __restrict__ cnt, const int* __restrict__ tok,
    bf16* __restrict__ mid) {
  const int e  = blockIdx.z;
  const int m0 = blockIdx.y * 64;
  const int n  = cnt[e];
  if (m0 >= n) return;

  const int wave = threadIdx.x >> 5;
  const int lane = threadIdx.x & 31;
  const int lm   = lane & 15;
  const int hf   = lane >> 4;
  const int gcol0 = blockIdx.x * 128 + wave * 16;

  __shared__ __align__(16) bf16 As[64][72];   // 64 rows x 64 (pad 8)
  __shared__ int toks[64];

  if (threadIdx.x < 64) {
    int m = m0 + threadIdx.x;
    toks[threadIdx.x] = (m < n) ? tok[e * CAPS + m] : -1;
  }
  __syncthreads();

  // per-lane weight row bases (B col = lm, K sub-offset = hf*16)
  const float* Wg = gup + ((size_t)e * (2 * I_DIM) + gcol0 + lm) * H_DIM + hf * 16;
  const float* Wu = Wg + (size_t)I_DIM * H_DIM;

  v8f accG[4] = {};
  v8f accU[4] = {};

  // A staging: thread -> (row = tid/4, 16 consecutive floats at (tid%4)*16)
  const int sr = threadIdx.x >> 2;
  const int sc = (threadIdx.x & 3) * 16;
  const int stk = toks[sr];
  const float* srcA = (stk >= 0) ? (hs + (size_t)stk * H_DIM + sc) : nullptr;

  for (int k0 = 0; k0 < H_DIM; k0 += 64) {
    // ---- stage A tile (64x64 fp32 -> bf16 in LDS) ----
    {
      float4 va = make_float4(0.f,0.f,0.f,0.f), vb = va, vc = va, vd = va;
      if (srcA) {
        const float4* p = (const float4*)(srcA + k0);
        va = p[0]; vb = p[1]; vc = p[2]; vd = p[3];
      }
      bf16* d = &As[sr][sc];
      d[0]=(bf16)va.x;  d[1]=(bf16)va.y;  d[2]=(bf16)va.z;  d[3]=(bf16)va.w;
      d[4]=(bf16)vb.x;  d[5]=(bf16)vb.y;  d[6]=(bf16)vb.z;  d[7]=(bf16)vb.w;
      d[8]=(bf16)vc.x;  d[9]=(bf16)vc.y;  d[10]=(bf16)vc.z; d[11]=(bf16)vc.w;
      d[12]=(bf16)vd.x; d[13]=(bf16)vd.y; d[14]=(bf16)vd.z; d[15]=(bf16)vd.w;
    }
    __syncthreads();

    #pragma unroll
    for (int ks = 0; ks < 2; ++ks) {
      // B fragments (gate + up): 16 consecutive fp32 per lane -> bf16
      const float* pg = Wg + k0 + ks * 32;
      const float* pu = Wu + k0 + ks * 32;
      v16bf bg, bu;
      #pragma unroll
      for (int j = 0; j < 16; ++j) { bg[j] = (bf16)pg[j]; bu[j] = (bf16)pu[j]; }
      __builtin_prefetch(pg + 64, 0, 1);
      __builtin_prefetch(pu + 64, 0, 1);
      // 4 m-subtiles reuse B
      #pragma unroll
      for (int mi = 0; mi < 4; ++mi) {
        v16bf a;
        const bf16* ar = &As[mi * 16 + lm][ks * 32 + hf * 16];
        #pragma unroll
        for (int j = 0; j < 8; ++j) { a[j] = ar[j]; a[8 + j] = ar[8 + j]; }
        accG[mi] = __builtin_amdgcn_wmma_f32_16x16x32_bf16(
            false, a, false, bg, (short)0, accG[mi], false, false);
        accU[mi] = __builtin_amdgcn_wmma_f32_16x16x32_bf16(
            false, a, false, bu, (short)0, accU[mi], false, false);
      }
    }
    __syncthreads();
  }

  // SiLU(gate) * up -> mid (bf16). c[r] -> row hf*8+r, col lm.
  #pragma unroll
  for (int mi = 0; mi < 4; ++mi) {
    #pragma unroll
    for (int r = 0; r < 8; ++r) {
      int m = mi * 16 + hf * 8 + r;
      float g = accG[mi][r], u = accU[mi][r];
      float s = g / (1.f + __expf(-g));
      mid[((size_t)e * CAPS + m0 + m) * I_DIM + gcol0 + lm] = (bf16)(s * u);
    }
  }
}

// ------------------------------------------------------------------
// Kernel 4: grouped GEMM2 (mid @ down^T), scale, atomic scatter-add.
// grid: (H_DIM/256, CAPS/64, R_EXP), block 256 (8 waves).
// wave w: 32 output cols [bx*256+w*32, +32), 4 m-subtiles.
// A staging uses gfx1250 async global->LDS DMA when available.
// ------------------------------------------------------------------
__global__ __launch_bounds__(256) void gemm2_kernel(
    const bf16* __restrict__ mid, const float* __restrict__ dwn,
    const int* __restrict__ cnt, const int* __restrict__ tok,
    const float* __restrict__ sw, float* __restrict__ out) {
  const int e  = blockIdx.z;
  const int m0 = blockIdx.y * 64;
  const int n  = cnt[e];
  if (m0 >= n) return;

  const int wave = threadIdx.x >> 5;
  const int lane = threadIdx.x & 31;
  const int lm   = lane & 15;
  const int hf   = lane >> 4;
  const int h0   = blockIdx.x * 256 + wave * 32;

  __shared__ __align__(16) bf16 As[64][72];
  __shared__ int   toks[64];
  __shared__ float sws[64];
  if (threadIdx.x < 64) {
    int m = m0 + threadIdx.x;
    toks[threadIdx.x] = (m < n) ? tok[e * CAPS + m] : 0;
    sws[threadIdx.x]  = (m < n) ? sw[e * CAPS + m] : 0.f;
  }
  __syncthreads();

  const float* W0 = dwn + ((size_t)e * H_DIM + h0 + lm) * I_DIM + hf * 16;      // cols h0+lm
  const float* W1 = W0 + (size_t)16 * I_DIM;                                    // cols h0+16+lm

  v8f acc0[4] = {};
  v8f acc1[4] = {};

  // A staging: thread -> (row = tid/4, 32 bytes at element (tid%4)*16)
  const int sr  = threadIdx.x >> 2;
  const int sc  = (threadIdx.x & 3) * 16;     // bf16 elements
  const bf16* srcA = mid + ((size_t)e * CAPS + m0 + sr) * I_DIM + sc;

  for (int k0 = 0; k0 < I_DIM; k0 += 64) {
#if USE_ASYNC_LDS
    __builtin_amdgcn_global_load_async_to_lds_b128(
        (gv4i*)(uintptr_t)(const void*)(srcA + k0),
        (lv4i*)(unsigned)(uintptr_t)&As[sr][sc], 0, 0);
    __builtin_amdgcn_global_load_async_to_lds_b128(
        (gv4i*)(uintptr_t)(const void*)(srcA + k0 + 8),
        (lv4i*)(unsigned)(uintptr_t)&As[sr][sc + 8], 0, 0);
    __builtin_amdgcn_s_wait_asynccnt(0);
#else
    {
      const uint4* p = (const uint4*)(srcA + k0);
      *(uint4*)&As[sr][sc]     = p[0];
      *(uint4*)&As[sr][sc + 8] = p[1];
    }
#endif
    __syncthreads();

    #pragma unroll
    for (int ks = 0; ks < 2; ++ks) {
      const float* p0 = W0 + k0 + ks * 32;
      const float* p1 = W1 + k0 + ks * 32;
      v16bf b0, b1;
      #pragma unroll
      for (int j = 0; j < 16; ++j) { b0[j] = (bf16)p0[j]; b1[j] = (bf16)p1[j]; }
      __builtin_prefetch(p0 + 64, 0, 1);
      __builtin_prefetch(p1 + 64, 0, 1);
      #pragma unroll
      for (int mi = 0; mi < 4; ++mi) {
        v16bf a;
        const bf16* ar = &As[mi * 16 + lm][ks * 32 + hf * 16];
        #pragma unroll
        for (int j = 0; j < 8; ++j) { a[j] = ar[j]; a[8 + j] = ar[8 + j]; }
        acc0[mi] = __builtin_amdgcn_wmma_f32_16x16x32_bf16(
            false, a, false, b0, (short)0, acc0[mi], false, false);
        acc1[mi] = __builtin_amdgcn_wmma_f32_16x16x32_bf16(
            false, a, false, b1, (short)0, acc1[mi], false, false);
      }
    }
    __syncthreads();
  }

  #pragma unroll
  for (int mi = 0; mi < 4; ++mi) {
    #pragma unroll
    for (int r = 0; r < 8; ++r) {
      int m = mi * 16 + hf * 8 + r;
      if (m0 + m < n) {
        float s = sws[m];
        float* dst = out + (size_t)toks[m] * H_DIM;
        unsafeAtomicAdd(dst + h0 + lm,      acc0[mi][r] * s);
        unsafeAtomicAdd(dst + h0 + 16 + lm, acc1[mi][r] * s);
      }
    }
  }
}

// ------------------------------------------------------------------
extern "C" void kernel_launch(void* const* d_in, const int* in_sizes, int n_in,
                              void* d_out, int out_size, void* d_ws, size_t ws_size,
                              hipStream_t stream) {
  const float*     hs  = (const float*)d_in[0];      // [T,H] f32
  const long long* idx = (const long long*)d_in[1];  // [T,K] i64
  const float*     tkw = (const float*)d_in[2];      // [T,K] f32
  const float*     gup = (const float*)d_in[3];      // [E,2I,H] f32
  const float*     dwn = (const float*)d_in[4];      // [R,H,I] f32
  float* out = (float*)d_out;                        // [T,H] f32

  char* ws = (char*)d_ws;
  int*   cnt = (int*)(ws + OFF_CNT);
  int*   tok = (int*)(ws + OFF_TOK);
  float* sw  = (float*)(ws + OFF_SW);
  float* zw  = (float*)(ws + OFF_ZW);
  bf16*  mid = (bf16*)(ws + OFF_MID);

  (void)hipMemsetAsync(cnt, 0, R_EXP * sizeof(int), stream);

  route_kernel<<<T_TOK / 256, 256, 0, stream>>>(idx, tkw, cnt, tok, sw, zw);
  zinit_kernel<<<(T_TOK * H_DIM) / 256, 256, 0, stream>>>(hs, zw, out);
  gemm1_kernel<<<dim3(I_DIM / 128, CAPS / 64, R_EXP), 256, 0, stream>>>(
      hs, gup, cnt, tok, mid);
  gemm2_kernel<<<dim3(H_DIM / 256, CAPS / 64, R_EXP), 256, 0, stream>>>(
      mid, dwn, cnt, tok, sw, out);
}
